// CausalPatternDetector_7275674599912
// MI455X (gfx1250) — matrix operand order repaired and checked by reading, not compiled, not run on hardware
//
#include <hip/hip_runtime.h>
#include <math.h>

// ---------------- problem constants ----------------
#define FDIM   1024          // features
#define TROWS  32767         // valid lagged rows (T - LAG)
#define TPAD   32768         // padded K (extra rows are zero)
#define F4     (FDIM / 4)
#define NKC    8             // split-K chunks
#define KCHUNK (TPAD / NKC)  // 4096 K per block

typedef __attribute__((ext_vector_type(16))) __bf16  v16bf;
typedef __attribute__((ext_vector_type(8)))  float   v8f;
typedef __attribute__((ext_vector_type(4)))  unsigned int v4u;

union FragU { v4u u[2]; v16bf b; };

// f32 -> bf16, round-to-nearest-even
__device__ __forceinline__ unsigned short f2bf(float f) {
    union { float f; unsigned u; } v; v.f = f;
    unsigned u = v.u;
    u += 0x7fffu + ((u >> 16) & 1u);
    return (unsigned short)(u >> 16);
}

// ---------------- pass 1: column sums over all 32768 rows ----------------
__global__ __launch_bounds__(256) void colsum_kernel(const float* __restrict__ x,
                                                     float* __restrict__ S) {
    const int tid = threadIdx.x;
    const int rb  = blockIdx.x;
    const float4* x4 = (const float4*)x;
    float4 s = make_float4(0.f, 0.f, 0.f, 0.f);
    const int r0 = rb * 256;
    for (int r = 0; r < 256; ++r) {
        float4 v = x4[(size_t)(r0 + r) * F4 + tid];
        s.x += v.x; s.y += v.y; s.z += v.z; s.w += v.w;
    }
    atomicAdd(&S[4 * tid + 0], s.x);
    atomicAdd(&S[4 * tid + 1], s.y);
    atomicAdd(&S[4 * tid + 2], s.z);
    atomicAdd(&S[4 * tid + 3], s.w);
}

// ---------------- pass 2: means ----------------
__global__ __launch_bounds__(256) void means_kernel(const float* __restrict__ x,
                                                    const float* __restrict__ S,
                                                    float* __restrict__ meanC,
                                                    float* __restrict__ meanL) {
    const int i = blockIdx.x * 256 + threadIdx.x;
    const float inv = 1.0f / (float)TROWS;
    meanC[i] = (S[i] - x[i]) * inv;
    meanL[i] = (S[i] - x[(size_t)TROWS * FDIM + i]) * inv;
}

// ---------------- fallback pass 3: column norms ----------------
__global__ __launch_bounds__(256) void norms_kernel(const float* __restrict__ x,
                                                    const float* __restrict__ meanC,
                                                    const float* __restrict__ meanL,
                                                    float* __restrict__ ncv,
                                                    float* __restrict__ nlv) {
    const int tid = threadIdx.x;
    const int rb  = blockIdx.x;
    const float4* x4 = (const float4*)x;
    const float4 mc = ((const float4*)meanC)[tid];
    const float4 ml = ((const float4*)meanL)[tid];
    float4 sc = make_float4(0.f, 0.f, 0.f, 0.f);
    float4 sl = make_float4(0.f, 0.f, 0.f, 0.f);
    const int r0 = rb * 256;
    for (int r = 0; r < 256; ++r) {
        int t = r0 + r;
        if (t >= TROWS) break;
        float4 a = x4[(size_t)(t + 1) * F4 + tid];
        float4 b = x4[(size_t)t * F4 + tid];
        float d;
        d = a.x - mc.x; sc.x += d * d;  d = a.y - mc.y; sc.y += d * d;
        d = a.z - mc.z; sc.z += d * d;  d = a.w - mc.w; sc.w += d * d;
        d = b.x - ml.x; sl.x += d * d;  d = b.y - ml.y; sl.y += d * d;
        d = b.z - ml.z; sl.z += d * d;  d = b.w - ml.w; sl.w += d * d;
    }
    atomicAdd(&ncv[4 * tid + 0], sc.x); atomicAdd(&ncv[4 * tid + 1], sc.y);
    atomicAdd(&ncv[4 * tid + 2], sc.z); atomicAdd(&ncv[4 * tid + 3], sc.w);
    atomicAdd(&nlv[4 * tid + 0], sl.x); atomicAdd(&nlv[4 * tid + 1], sl.y);
    atomicAdd(&nlv[4 * tid + 2], sl.z); atomicAdd(&nlv[4 * tid + 3], sl.w);
}

// ---------------- fast pass 3: center + bf16 + transpose + norms -----------
#define TSTR 72

__global__ __launch_bounds__(256) void prep_kernel(const float* __restrict__ x,
                                                   const float* __restrict__ meanC,
                                                   const float* __restrict__ meanL,
                                                   float* __restrict__ ncv,
                                                   float* __restrict__ nlv,
                                                   unsigned short* __restrict__ ZcT,
                                                   unsigned short* __restrict__ ZlT) {
    __shared__ alignas(16) unsigned short tC[64 * TSTR];
    __shared__ alignas(16) unsigned short tL[64 * TSTR];
    __shared__ float redC[64];
    __shared__ float redL[64];

    const int tid = threadIdx.x;
    const int c0  = blockIdx.x * 64;
    const int t0  = blockIdx.y * 1024;
    const int tc  = tid & 15;
    const int tr  = tid >> 4;
    const int cl  = 4 * tc;

    const float4* x4 = (const float4*)x;
    const float4 mc = ((const float4*)(meanC + c0))[tc];
    const float4 ml = ((const float4*)(meanL + c0))[tc];

    if (tid < 64) { redC[tid] = 0.f; redL[tid] = 0.f; }

    float pc0 = 0.f, pc1 = 0.f, pc2 = 0.f, pc3 = 0.f;
    float pl0 = 0.f, pl1 = 0.f, pl2 = 0.f, pl3 = 0.f;

    const int crow = tid >> 2;
    const int tq   = (tid & 3) * 16;

    for (int sub = 0; sub < 16; ++sub) {
        __syncthreads();
#pragma unroll
        for (int p = 0; p < 4; ++p) {
            const int ts = sub * 64 + p * 16 + tr;
            const int t  = t0 + ts;
            unsigned short a0 = 0, a1 = 0, a2 = 0, a3 = 0;
            unsigned short b0 = 0, b1 = 0, b2 = 0, b3 = 0;
            if (t < TROWS) {
                float4 av = x4[(size_t)(t + 1) * F4 + (c0 >> 2) + tc];
                float4 bv = x4[(size_t)t * F4 + (c0 >> 2) + tc];
                float d;
                d = av.x - mc.x; pc0 += d * d; a0 = f2bf(d);
                d = av.y - mc.y; pc1 += d * d; a1 = f2bf(d);
                d = av.z - mc.z; pc2 += d * d; a2 = f2bf(d);
                d = av.w - mc.w; pc3 += d * d; a3 = f2bf(d);
                d = bv.x - ml.x; pl0 += d * d; b0 = f2bf(d);
                d = bv.y - ml.y; pl1 += d * d; b1 = f2bf(d);
                d = bv.z - ml.z; pl2 += d * d; b2 = f2bf(d);
                d = bv.w - ml.w; pl3 += d * d; b3 = f2bf(d);
            }
            const int tsub = p * 16 + tr;
            tC[(cl + 0) * TSTR + tsub] = a0;
            tC[(cl + 1) * TSTR + tsub] = a1;
            tC[(cl + 2) * TSTR + tsub] = a2;
            tC[(cl + 3) * TSTR + tsub] = a3;
            tL[(cl + 0) * TSTR + tsub] = b0;
            tL[(cl + 1) * TSTR + tsub] = b1;
            tL[(cl + 2) * TSTR + tsub] = b2;
            tL[(cl + 3) * TSTR + tsub] = b3;
        }
        __syncthreads();
        const size_t gbase = (size_t)(c0 + crow) * TPAD + t0 + sub * 64 + tq;
        v4u* dC = (v4u*)(ZcT + gbase);
        v4u* dL = (v4u*)(ZlT + gbase);
        const v4u* sC = (const v4u*)&tC[crow * TSTR + tq];
        const v4u* sL = (const v4u*)&tL[crow * TSTR + tq];
        dC[0] = sC[0]; dC[1] = sC[1];
        dL[0] = sL[0]; dL[1] = sL[1];
    }

    atomicAdd(&redC[cl + 0], pc0); atomicAdd(&redC[cl + 1], pc1);
    atomicAdd(&redC[cl + 2], pc2); atomicAdd(&redC[cl + 3], pc3);
    atomicAdd(&redL[cl + 0], pl0); atomicAdd(&redL[cl + 1], pl1);
    atomicAdd(&redL[cl + 2], pl2); atomicAdd(&redL[cl + 3], pl3);
    __syncthreads();
    if (tid < 64) {
        atomicAdd(&ncv[c0 + tid], redC[tid]);
        atomicAdd(&nlv[c0 + tid], redL[tid]);
    }
}

// ---------------- fast pass 4: split-K WMMA GEMM, async-LDS double-buffered --
#define KCH  128   // K per stage
#define KSTR 136   // LDS row stride (bf16): 272B, 16B-aligned, bank-skewed
#define NST  (KCHUNK / KCH)   // 32 stages per block

__global__ __launch_bounds__(256) void gemm_part_kernel(
        const unsigned short* __restrict__ ZcT,
        const unsigned short* __restrict__ ZlT,
        float* __restrict__ covp) {
    __shared__ alignas(16) unsigned short ldsA[2][128 * KSTR];
    __shared__ alignas(16) unsigned short ldsB[2][128 * KSTR];

    const int tid = threadIdx.x;
    const int i0 = blockIdx.y * 128;
    const int j0 = blockIdx.x * 128;
    const int kc = blockIdx.z;          // split-K chunk
    const int k0 = kc * KCHUNK;

    const int lane = tid & 31;
    const int w    = tid >> 5;
    const int wm   = w & 3;
    const int wn   = w >> 2;
    const int l16  = lane & 15;
    const int lh   = lane >> 4;

    v8f acc[2][4];
#pragma unroll
    for (int mt = 0; mt < 2; ++mt)
#pragma unroll
        for (int nt = 0; nt < 4; ++nt)
            acc[mt][nt] = (v8f){0.f, 0.f, 0.f, 0.f, 0.f, 0.f, 0.f, 0.f};

    // staging map: 2 threads per column, 64 K elems (128B) each
    const int scol = tid >> 1;
    const int skh  = tid & 1;
    const unsigned long long gA0 =
        (unsigned long long)(ZcT + (size_t)(i0 + scol) * TPAD + k0 + 64 * skh);
    const unsigned long long gB0 =
        (unsigned long long)(ZlT + (size_t)(j0 + scol) * TPAD + k0 + 64 * skh);
    unsigned laoff[2], lboff[2];
    laoff[0] = (unsigned)(unsigned long long)&ldsA[0][scol * KSTR + 64 * skh];
    laoff[1] = (unsigned)(unsigned long long)&ldsA[1][scol * KSTR + 64 * skh];
    lboff[0] = (unsigned)(unsigned long long)&ldsB[0][scol * KSTR + 64 * skh];
    lboff[1] = (unsigned)(unsigned long long)&ldsB[1][scol * KSTR + 64 * skh];

    // kick stage 0 into buffer 0 (CDNA5 async global->LDS, ASYNCcnt-tracked;
    // INST_OFFSET applies to both the global and the LDS address)
#define ISSUE_STAGE(BUF, GOFF)                                                  \
    {                                                                           \
        unsigned long long ga = gA0 + (GOFF);                                   \
        unsigned long long gb = gB0 + (GOFF);                                   \
        unsigned la = laoff[BUF], lb = lboff[BUF];                              \
        asm volatile("global_load_async_to_lds_b128 %0, %1, off offset:0"  :: "v"(la), "v"(ga) : "memory"); \
        asm volatile("global_load_async_to_lds_b128 %0, %1, off offset:16" :: "v"(la), "v"(ga) : "memory"); \
        asm volatile("global_load_async_to_lds_b128 %0, %1, off offset:32" :: "v"(la), "v"(ga) : "memory"); \
        asm volatile("global_load_async_to_lds_b128 %0, %1, off offset:48" :: "v"(la), "v"(ga) : "memory"); \
        asm volatile("global_load_async_to_lds_b128 %0, %1, off offset:64" :: "v"(la), "v"(ga) : "memory"); \
        asm volatile("global_load_async_to_lds_b128 %0, %1, off offset:80" :: "v"(la), "v"(ga) : "memory"); \
        asm volatile("global_load_async_to_lds_b128 %0, %1, off offset:96" :: "v"(la), "v"(ga) : "memory"); \
        asm volatile("global_load_async_to_lds_b128 %0, %1, off offset:112" :: "v"(la), "v"(ga) : "memory"); \
        asm volatile("global_load_async_to_lds_b128 %0, %1, off offset:0"  :: "v"(lb), "v"(gb) : "memory"); \
        asm volatile("global_load_async_to_lds_b128 %0, %1, off offset:16" :: "v"(lb), "v"(gb) : "memory"); \
        asm volatile("global_load_async_to_lds_b128 %0, %1, off offset:32" :: "v"(lb), "v"(gb) : "memory"); \
        asm volatile("global_load_async_to_lds_b128 %0, %1, off offset:48" :: "v"(lb), "v"(gb) : "memory"); \
        asm volatile("global_load_async_to_lds_b128 %0, %1, off offset:64" :: "v"(lb), "v"(gb) : "memory"); \
        asm volatile("global_load_async_to_lds_b128 %0, %1, off offset:80" :: "v"(lb), "v"(gb) : "memory"); \
        asm volatile("global_load_async_to_lds_b128 %0, %1, off offset:96" :: "v"(lb), "v"(gb) : "memory"); \
        asm volatile("global_load_async_to_lds_b128 %0, %1, off offset:112" :: "v"(lb), "v"(gb) : "memory"); \
    }

    ISSUE_STAGE(0, 0ull)

    for (int s = 0; s < NST; ++s) {
        const int buf = s & 1;
        asm volatile("s_wait_asynccnt 0x0" ::: "memory");
        __syncthreads();
        if (s + 1 < NST) {
            ISSUE_STAGE((s + 1) & 1, (unsigned long long)((s + 1) * KCH * 2))
        }
        const unsigned short* lA = ldsA[buf];
        const unsigned short* lB = ldsB[buf];
#pragma unroll
        for (int ks = 0; ks < KCH; ks += 32) {
            FragU a[2], b[4];
#pragma unroll
            for (int mt = 0; mt < 2; ++mt) {
                const int r = 32 * wm + 16 * mt + l16;
                const unsigned short* base = &lA[r * KSTR + ks + 8 * lh];
                a[mt].u[0] = *(const v4u*)base;
                a[mt].u[1] = *(const v4u*)(base + 16);
            }
#pragma unroll
            for (int nt = 0; nt < 4; ++nt) {
                const int c = 64 * wn + 16 * nt + l16;
                const unsigned short* base = &lB[c * KSTR + ks + 8 * lh];
                b[nt].u[0] = *(const v4u*)base;
                b[nt].u[1] = *(const v4u*)(base + 16);
            }
#pragma unroll
            for (int mt = 0; mt < 2; ++mt)
#pragma unroll
                for (int nt = 0; nt < 4; ++nt)
                    acc[mt][nt] = __builtin_amdgcn_wmma_f32_16x16x32_bf16(
                        false, a[mt].b, false, b[nt].b,
                        (short)0, acc[mt][nt], false, false);
        }
    }
#undef ISSUE_STAGE

    // store partial cov slab, transposed layout covp[kc][j][i] (all slots written)
    float* slab = covp + ((size_t)kc << 20);
#pragma unroll
    for (int mt = 0; mt < 2; ++mt) {
        const int ibase = i0 + 32 * wm + 16 * mt + 8 * lh;
#pragma unroll
        for (int nt = 0; nt < 4; ++nt) {
            const int j = j0 + 64 * wn + 16 * nt + l16;
            float4 lo, hi;
            lo.x = acc[mt][nt][0]; lo.y = acc[mt][nt][1];
            lo.z = acc[mt][nt][2]; lo.w = acc[mt][nt][3];
            hi.x = acc[mt][nt][4]; hi.y = acc[mt][nt][5];
            hi.z = acc[mt][nt][6]; hi.w = acc[mt][nt][7];
            float4* dst = (float4*)(slab + (size_t)j * FDIM + ibase);
            dst[0] = lo;
            dst[1] = hi;
        }
    }
}

// ---------------- fast pass 5: reduce slabs -> corr sum ----------------
__global__ __launch_bounds__(256) void reduce_corr_kernel(
        const float* __restrict__ covp,
        const float* __restrict__ ncv,
        const float* __restrict__ nlv,
        float* __restrict__ out) {
    __shared__ float wsum[8];
    const int tid = threadIdx.x;
    const int j   = blockIdx.x;
    const float nlj = nlv[j];
    const float4 nci = ((const float4*)ncv)[tid];

    float4 s = make_float4(0.f, 0.f, 0.f, 0.f);
#pragma unroll
    for (int c = 0; c < NKC; ++c) {
        float4 v = *(const float4*)(covp + ((size_t)c << 20) + (size_t)j * FDIM + 4 * tid);
        s.x += v.x; s.y += v.y; s.z += v.z; s.w += v.w;
    }
    const int i = 4 * tid;
    float sum = 0.f;
    if (i + 0 != j) sum += fabsf(s.x) * rsqrtf(nci.x * nlj);
    if (i + 1 != j) sum += fabsf(s.y) * rsqrtf(nci.y * nlj);
    if (i + 2 != j) sum += fabsf(s.z) * rsqrtf(nci.z * nlj);
    if (i + 3 != j) sum += fabsf(s.w) * rsqrtf(nci.w * nlj);

#pragma unroll
    for (int o = 16; o > 0; o >>= 1) sum += __shfl_xor(sum, o, 32);
    if ((tid & 31) == 0) wsum[tid >> 5] = sum;
    __syncthreads();
    if (tid == 0) {
        float t = 0.f;
#pragma unroll
        for (int q = 0; q < 8; ++q) t += wsum[q];
        atomicAdd(out, t * (1.0f / ((float)FDIM * (float)(FDIM - 1))));
    }
}

// ---------------- fallback pass 4: fused f32->bf16 WMMA GEMM ----------------
#define KSTRF 72

__global__ __launch_bounds__(256) void gemm_corr_kernel(const float* __restrict__ x,
                                                        const float* __restrict__ meanC,
                                                        const float* __restrict__ meanL,
                                                        const float* __restrict__ ncv,
                                                        const float* __restrict__ nlv,
                                                        float* __restrict__ out) {
    __shared__ alignas(16) unsigned short ldsA[128 * KSTRF];
    __shared__ alignas(16) unsigned short ldsB[128 * KSTRF];
    __shared__ float mC[128];
    __shared__ float mL[128];

    const int tid = threadIdx.x;
    const int i0 = blockIdx.y * 128;
    const int j0 = blockIdx.x * 128;

    if (tid < 128) mC[tid] = meanC[i0 + tid];
    else           mL[tid - 128] = meanL[j0 + tid - 128];

    const int lane = tid & 31;
    const int w    = tid >> 5;
    const int wm   = w & 3;
    const int wn   = w >> 2;
    const int l16  = lane & 15;
    const int lh   = lane >> 4;

    v8f acc[2][4];
#pragma unroll
    for (int mt = 0; mt < 2; ++mt)
#pragma unroll
        for (int nt = 0; nt < 4; ++nt)
            acc[mt][nt] = (v8f){0.f, 0.f, 0.f, 0.f, 0.f, 0.f, 0.f, 0.f};

    const int i4 = tid & 31;
    const int tl = tid >> 5;
    const float4* x4 = (const float4*)x;

    for (int kb = 0; kb < TPAD; kb += 64) {
        __syncthreads();
#pragma unroll
        for (int p = 0; p < 8; ++p) {
            const int kl = tl + p * 8;
            const int t  = kb + kl;
            const int ci = 4 * i4;
            unsigned short za0 = 0, za1 = 0, za2 = 0, za3 = 0;
            unsigned short zb0 = 0, zb1 = 0, zb2 = 0, zb3 = 0;
            if (t < TROWS) {
                float4 av = x4[(size_t)(t + 1) * F4 + (i0 >> 2) + i4];
                float4 bv = x4[(size_t)t * F4 + (j0 >> 2) + i4];
                za0 = f2bf(av.x - mC[ci + 0]); za1 = f2bf(av.y - mC[ci + 1]);
                za2 = f2bf(av.z - mC[ci + 2]); za3 = f2bf(av.w - mC[ci + 3]);
                zb0 = f2bf(bv.x - mL[ci + 0]); zb1 = f2bf(bv.y - mL[ci + 1]);
                zb2 = f2bf(bv.z - mL[ci + 2]); zb3 = f2bf(bv.w - mL[ci + 3]);
            }
            ldsA[(ci + 0) * KSTRF + kl] = za0;
            ldsA[(ci + 1) * KSTRF + kl] = za1;
            ldsA[(ci + 2) * KSTRF + kl] = za2;
            ldsA[(ci + 3) * KSTRF + kl] = za3;
            ldsB[(ci + 0) * KSTRF + kl] = zb0;
            ldsB[(ci + 1) * KSTRF + kl] = zb1;
            ldsB[(ci + 2) * KSTRF + kl] = zb2;
            ldsB[(ci + 3) * KSTRF + kl] = zb3;
        }
        __syncthreads();

#pragma unroll
        for (int ks = 0; ks < 64; ks += 32) {
            FragU a[2], b[4];
#pragma unroll
            for (int mt = 0; mt < 2; ++mt) {
                const int r = 32 * wm + 16 * mt + l16;
                const unsigned short* base = &ldsA[r * KSTRF + ks + 8 * lh];
                a[mt].u[0] = *(const v4u*)base;
                a[mt].u[1] = *(const v4u*)(base + 16);
            }
#pragma unroll
            for (int nt = 0; nt < 4; ++nt) {
                const int c = 64 * wn + 16 * nt + l16;
                const unsigned short* base = &ldsB[c * KSTRF + ks + 8 * lh];
                b[nt].u[0] = *(const v4u*)base;
                b[nt].u[1] = *(const v4u*)(base + 16);
            }
#pragma unroll
            for (int mt = 0; mt < 2; ++mt)
#pragma unroll
                for (int nt = 0; nt < 4; ++nt)
                    acc[mt][nt] = __builtin_amdgcn_wmma_f32_16x16x32_bf16(
                        false, a[mt].b, false, b[nt].b,
                        (short)0, acc[mt][nt], false, false);
        }
    }

    float sum = 0.f;
#pragma unroll
    for (int mt = 0; mt < 2; ++mt) {
        const int ibase = i0 + 32 * wm + 16 * mt + 8 * lh;
#pragma unroll
        for (int nt = 0; nt < 4; ++nt) {
            const int j = j0 + 64 * wn + 16 * nt + l16;
            const float nlj = nlv[j];
#pragma unroll
            for (int v = 0; v < 8; ++v) {
                const int i = ibase + v;
                const float c = acc[mt][nt][v];
                if (i != j) sum += fabsf(c) * rsqrtf(ncv[i] * nlj);
            }
        }
    }
#pragma unroll
    for (int o = 16; o > 0; o >>= 1) sum += __shfl_xor(sum, o, 32);
    if (lane == 0)
        atomicAdd(out, sum * (1.0f / ((float)FDIM * (float)(FDIM - 1))));
}

// ---------------- launch ----------------
extern "C" void kernel_launch(void* const* d_in, const int* in_sizes, int n_in,
                              void* d_out, int out_size, void* d_ws, size_t ws_size,
                              hipStream_t stream) {
    (void)in_sizes; (void)n_in; (void)out_size;
    const float* x = (const float*)d_in[0];
    float* ws    = (float*)d_ws;
    float* S     = ws;              // 1024
    float* meanC = ws + 1024;       // 1024
    float* meanL = ws + 2048;       // 1024
    float* ncv   = ws + 3072;       // 1024
    float* nlv   = ws + 4096;       // 1024

    hipMemsetAsync(d_ws, 0, 5120 * sizeof(float), stream);
    hipMemsetAsync(d_out, 0, sizeof(float), stream);

    colsum_kernel<<<128, 256, 0, stream>>>(x, S);
    means_kernel<<<4, 256, 0, stream>>>(x, S, meanC, meanL);

    const size_t panel = (size_t)TPAD * FDIM;   // bf16 elems per operand
    const size_t covfl = (size_t)NKC * FDIM * FDIM;  // partial-cov floats
    const size_t need  = 5120 * sizeof(float)
                       + 2 * panel * sizeof(unsigned short)
                       + covfl * sizeof(float);
    if (ws_size >= need) {
        unsigned short* ZcT = (unsigned short*)(ws + 5120);
        unsigned short* ZlT = ZcT + panel;
        float* covp = (float*)(ZlT + panel);
        prep_kernel<<<dim3(16, 32), 256, 0, stream>>>(x, meanC, meanL, ncv, nlv,
                                                      ZcT, ZlT);
        gemm_part_kernel<<<dim3(8, 8, NKC), 256, 0, stream>>>(ZcT, ZlT, covp);
        reduce_corr_kernel<<<FDIM, 256, 0, stream>>>(covp, ncv, nlv,
                                                     (float*)d_out);
    } else {
        norms_kernel<<<128, 256, 0, stream>>>(x, meanC, meanL, ncv, nlv);
        gemm_corr_kernel<<<dim3(8, 8), 256, 0, stream>>>(x, meanC, meanL, ncv, nlv,
                                                         (float*)d_out);
    }
}
